// MambaModel_5351529250913
// MI455X (gfx1250) — matrix operand order repaired and checked by reading, not compiled, not run on hardware
//
#include <hip/hip_runtime.h>
#include <hip/hip_bf16.h>
#include <cstdint>
#include <cstddef>

// ---------------------------------------------------------------------------
// Mamba block forward for MI455X (gfx1250, wave32, WMMA + async-LDS staging).
// Batch is a broadcast of the same learned input -> compute one batch's
// (1024 x ...) block and duplicate the final output into both batch slices.
// ---------------------------------------------------------------------------

typedef __attribute__((ext_vector_type(16))) _Float16 v16h;
typedef __attribute__((ext_vector_type(8)))  float    v8f;
typedef __attribute__((ext_vector_type(4)))  float    f4;

#define SEQ     1024
#define DMODEL  2048
#define DINNER  4096
#define DTRANK  128
#define DSTATE  16
#define DBCW    160      // DTRANK + 2*DSTATE
#define DOUTN   1024

#if __has_builtin(__builtin_amdgcn_sched_barrier)
#define SCHED_FENCE() __builtin_amdgcn_sched_barrier(0)
#else
#define SCHED_FENCE() asm volatile("" ::: "memory")
#endif

__device__ __forceinline__ float silu_f(float x) {
    return x * (1.0f / (1.0f + __expf(-x)));
}

// --- CDNA5 async global->LDS copy (ASYNCcnt), ISA §15.18 opcode 98 ----------
__device__ __forceinline__ void async_copy_b128(uint32_t ldsAddr, const float* gaddr) {
    asm volatile("global_load_async_to_lds_b128 %0, %1, off"
                 :: "v"(ldsAddr), "v"(gaddr) : "memory");
}
__device__ __forceinline__ void wait_async0() {
    asm volatile("s_wait_asynccnt 0x0" ::: "memory");
}
__device__ __forceinline__ void wait_async1() {
    asm volatile("s_wait_asynccnt 0x1" ::: "memory");
}

__device__ __forceinline__ v16h pack16(f4 a0, f4 a1, f4 a2, f4 a3) {
    v16h r;
    r[0]  = (_Float16)a0.x; r[1]  = (_Float16)a0.y;
    r[2]  = (_Float16)a0.z; r[3]  = (_Float16)a0.w;
    r[4]  = (_Float16)a1.x; r[5]  = (_Float16)a1.y;
    r[6]  = (_Float16)a1.z; r[7]  = (_Float16)a1.w;
    r[8]  = (_Float16)a2.x; r[9]  = (_Float16)a2.y;
    r[10] = (_Float16)a2.z; r[11] = (_Float16)a2.w;
    r[12] = (_Float16)a3.x; r[13] = (_Float16)a3.y;
    r[14] = (_Float16)a3.z; r[15] = (_Float16)a3.w;
    return r;
}

// ---------------------------------------------------------------------------
// WMMA GEMM:  C[M,N] = A[M,K] * W[N,K]^T  (row-major fp32 in, f16 WMMA,
// f32 accumulate).  Block = 4 waves, tile = 16(M) x 256(N), K step 32.
// A chunk (16x32) is async-staged to LDS (double buffered) and its fragment
// is reused by every wave across 4 N-subtiles.  Inner step is phase-split
// with sched_barriers: [16 b128 loads] | [32 cvts] | [4 wmmas], so all
// loads are in flight together and wmmas issue back-to-back.
// FULL  : N is a multiple of 256 -> no subtile guards at all.
// !FULL : guards evaluated on SGPRs (readfirstlane) -> s_cbranch, no exec mask.
// MODE 0: plain store
// MODE 1: softplus(acc + bias[col])        (delta path)
// MODE 2: acc + bias[col], stored twice    (final output, batch duplicate)
// ---------------------------------------------------------------------------
template <int MODE, bool FULL>
__global__ __launch_bounds__(128) void wmma_gemm_tile(
    const float* __restrict__ A, const float* __restrict__ W,
    float* __restrict__ C, int N, int K, int lda, int ldw, int ldc,
    const float* __restrict__ bias, long long dup)
{
    __shared__ float sA[2][16][36];          // padded stride to spread LDS banks

    const int tid   = threadIdx.x;           // 0..127
    const int lane  = tid & 31;
    const int half  = lane >> 4;             // 0|1
    const int l15   = lane & 15;
    const int tileM = blockIdx.y << 4;
    const int blkN  = blockIdx.x << 8;       // 256 columns per block

    // wave id as an SGPR value (uniform within the wave by construction)
    const int swid  = __builtin_amdgcn_readfirstlane(tid >> 5);   // 0..3

    // ---- async staging: each thread owns one 16B piece of the 16x32 A chunk
    const int arow = tid >> 3;               // 0..15
    const int acol = (tid & 7) << 2;         // 0,4,...,28
    const float* aG = A + (size_t)(tileM + arow) * lda + acol;
    const uint32_t ldsDst[2] = {
        (uint32_t)(uintptr_t)&sA[0][arow][acol],
        (uint32_t)(uintptr_t)&sA[1][arow][acol]
    };

    // ---- per-wave N subtiles; nt/ntv are SGPR-derived => scalar branches
    int  nt[4];
    bool ntv[4];
    const float* bp[4];
#pragma unroll
    for (int j = 0; j < 4; ++j) {
        nt[j]  = blkN + (swid << 6) + (j << 4);
        ntv[j] = FULL ? true : (nt[j] < N);
        bp[j]  = W + (size_t)(nt[j] + l15) * ldw + (half << 4);
    }

    v8f acc[4] = {{}, {}, {}, {}};

    async_copy_b128(ldsDst[0], aG);          // prefetch chunk 0
    int buf = 0;

    for (int k = 0; k < K; k += 32) {
        const bool more = (k + 32) < K;
        if (more) async_copy_b128(ldsDst[buf ^ 1], aG + k + 32);
        if (more) wait_async1(); else wait_async0();   // chunk k landed (in-order)
        __syncthreads();                                // visible to all waves

        // A fragment (16x32 f16 layout) from LDS
        const float* ar = &sA[buf][l15][half << 4];
        const v16h af = pack16(*(const f4*)(ar),      *(const f4*)(ar + 4),
                               *(const f4*)(ar + 8),  *(const f4*)(ar + 12));

        // Phase 1: issue ALL B-fragment loads into distinct buffers
        f4 br[4][4];
#pragma unroll
        for (int j = 0; j < 4; ++j) {
            if (FULL || ntv[j]) {
                const float* b = bp[j] + k;
                br[j][0] = *(const f4*)(b);
                br[j][1] = *(const f4*)(b + 4);
                br[j][2] = *(const f4*)(b + 8);
                br[j][3] = *(const f4*)(b + 12);
            }
        }
        SCHED_FENCE();                       // loads may not sink below here

        // Phase 2: convert all fragments (partial loadcnt waits as loads drain)
        v16h bf[4];
#pragma unroll
        for (int j = 0; j < 4; ++j) {
            if (FULL || ntv[j])
                bf[j] = pack16(br[j][0], br[j][1], br[j][2], br[j][3]);
        }
        SCHED_FENCE();                       // cvts may not sink into wmma phase

        // Phase 3: 4 independent wmmas back-to-back (no RAW between them)
#pragma unroll
        for (int j = 0; j < 4; ++j) {
            if (FULL || ntv[j]) {
                acc[j] = __builtin_amdgcn_wmma_f32_16x16x32_f16(
                    false, af, false, bf[j], (short)0, acc[j], false, false);
            }
        }
        __syncthreads();                     // all waves done with buf before reissue
        buf ^= 1;
    }

    // ---- store: C/D layout -> vgpr r holds (M = r + 8*half, N = l15)
    const int rb = half << 3;
#pragma unroll
    for (int j = 0; j < 4; ++j) {
        if (!(FULL || ntv[j])) continue;
        const int col = nt[j] + l15;
#pragma unroll
        for (int r = 0; r < 8; ++r) {
            const int row = tileM + rb + r;
            float v = acc[j][r];
            const size_t off = (size_t)row * ldc + col;
            if (MODE == 0) {
                C[off] = v;
            } else if (MODE == 1) {
                v += bias[col];
                C[off] = (v > 20.0f) ? v : __logf(1.0f + __expf(v));
            } else {
                v += bias[col];
                C[off] = v;
                C[off + (size_t)dup] = v;
            }
        }
    }
}

// ---------------------------------------------------------------------------
// Causal depthwise conv1d (kernel 4) + bias + SiLU.  xs = xz[:, 0:DINNER].
// ---------------------------------------------------------------------------
__global__ __launch_bounds__(256) void conv_silu_kernel(
    const float* __restrict__ xz, const float* __restrict__ cw,
    const float* __restrict__ cb, float* __restrict__ u)
{
    const int idx = blockIdx.x * 256 + threadIdx.x;   // over SEQ*DINNER
    const int t = idx >> 12;
    const int d = idx & (DINNER - 1);
    float acc = cb[d];
    const float* w = cw + d * 4;
#pragma unroll
    for (int i = 0; i < 4; ++i) {
        const int ts = t - 3 + i;
        if (ts >= 0)
            acc = fmaf(w[i], xz[(size_t)ts * (2 * DINNER) + d], acc);
    }
    u[idx] = silu_f(acc);
}

// ---------------------------------------------------------------------------
// Selective scan: one thread per channel d; h[16] in registers.
// B_t/C_t (32 floats per timestep) staged in LDS in 32-step chunks.
// ---------------------------------------------------------------------------
__global__ __launch_bounds__(256) void scan_kernel(
    const float* __restrict__ delta, const float* __restrict__ u,
    const float* __restrict__ dbc, const float* __restrict__ A_log,
    float* __restrict__ ys)
{
    __shared__ float sBC[32][32];   // [t_local][0..15]=B, [16..31]=C
    const int d = blockIdx.x * 256 + threadIdx.x;

    float Arow[DSTATE];
#pragma unroll
    for (int s = 0; s < DSTATE; ++s)
        Arow[s] = -__expf(A_log[(size_t)d * DSTATE + s]);

    float h[DSTATE];
#pragma unroll
    for (int s = 0; s < DSTATE; ++s) h[s] = 0.0f;

    for (int tc = 0; tc < SEQ; tc += 32) {
#pragma unroll
        for (int i = 0; i < 4; ++i) {
            const int idx = threadIdx.x + (i << 8);   // 0..1023
            const int tl = idx >> 5, j = idx & 31;
            sBC[tl][j] = dbc[(size_t)(tc + tl) * DBCW + DTRANK + j];
        }
        __syncthreads();

        for (int tl = 0; tl < 32; ++tl) {
            const int t = tc + tl;
            const float dlt = delta[(size_t)t * DINNER + d];
            const float du  = dlt * u[(size_t)t * DINNER + d];
            float y = 0.0f;
#pragma unroll
            for (int s = 0; s < DSTATE; ++s) {
                const float e = __expf(dlt * Arow[s]);
                h[s] = fmaf(e, h[s], du * sBC[tl][s]);
                y = fmaf(h[s], sBC[tl][16 + s], y);
            }
            ys[(size_t)t * DINNER + d] = y;
        }
        __syncthreads();
    }
}

// ---------------------------------------------------------------------------
// y = (ys + u * D_skip[d]) * silu(z),  z = xz[:, DINNER:2*DINNER].  In-place.
// ---------------------------------------------------------------------------
__global__ __launch_bounds__(256) void combine_kernel(
    const float* __restrict__ xz, const float* __restrict__ u,
    const float* __restrict__ D_skip, float* __restrict__ ys)
{
    const int idx = blockIdx.x * 256 + threadIdx.x;
    const int t = idx >> 12;
    const int d = idx & (DINNER - 1);
    const float z = xz[(size_t)t * (2 * DINNER) + DINNER + d];
    ys[idx] = (ys[idx] + u[idx] * D_skip[d]) * silu_f(z);
}

// ---------------------------------------------------------------------------
extern "C" void kernel_launch(void* const* d_in, const int* in_sizes, int n_in,
                              void* d_out, int out_size, void* d_ws, size_t ws_size,
                              hipStream_t stream)
{
    const float* x        = (const float*)d_in[0];   // (1,1024,2048)
    const float* in_proj  = (const float*)d_in[1];   // (8192,2048)
    const float* conv_w   = (const float*)d_in[2];   // (4096,4)
    const float* conv_b   = (const float*)d_in[3];   // (4096,)
    const float* x_proj   = (const float*)d_in[4];   // (160,4096)
    const float* dt_proj  = (const float*)d_in[5];   // (4096,128)
    const float* dt_b     = (const float*)d_in[6];   // (4096,)
    const float* A_log    = (const float*)d_in[7];   // (4096,16)
    const float* D_skip   = (const float*)d_in[8];   // (4096,)
    const float* out_proj = (const float*)d_in[9];   // (2048,4096)
    const float* out_w    = (const float*)d_in[10];  // (1024,2048)
    const float* out_b    = (const float*)d_in[11];  // (1024,)
    float* out = (float*)d_out;                      // (2,1024,1024)

    // Workspace layout (floats), ~93 MB total.
    float* ws    = (float*)d_ws;
    float* xz    = ws;                                  // 1024*8192
    float* u     = xz    + (size_t)SEQ * 2 * DINNER;    // 1024*4096
    float* dbc   = u     + (size_t)SEQ * DINNER;        // 1024*160
    float* delta = dbc   + (size_t)SEQ * DBCW;          // 1024*4096
    float* ys    = delta + (size_t)SEQ * DINNER;        // 1024*4096
    float* o1    = ys    + (size_t)SEQ * DINNER;        // 1024*2048

    const dim3 blk(128);
    auto grd = [](int N) { return dim3((N + 255) / 256, SEQ / 16); };

    // 1) xz = x @ in_proj^T                      (1024 x 8192, K=2048)
    wmma_gemm_tile<0, true><<<grd(2 * DINNER), blk, 0, stream>>>(
        x, in_proj, xz, 2 * DINNER, DMODEL, DMODEL, DMODEL, 2 * DINNER,
        nullptr, 0);

    // 2) u = silu(causal depthwise conv(xs) + b)
    conv_silu_kernel<<<SEQ * DINNER / 256, 256, 0, stream>>>(xz, conv_w, conv_b, u);

    // 3) dbc = u @ x_proj^T                      (1024 x 160, K=4096, partial N)
    wmma_gemm_tile<0, false><<<grd(DBCW), blk, 0, stream>>>(
        u, x_proj, dbc, DBCW, DINNER, DINNER, DINNER, DBCW, nullptr, 0);

    // 4) delta = softplus(dt @ dt_proj^T + b)    (1024 x 4096, K=128; A=dbc lda=160)
    wmma_gemm_tile<1, true><<<grd(DINNER), blk, 0, stream>>>(
        dbc, dt_proj, delta, DINNER, DTRANK, DBCW, DTRANK, DINNER, dt_b, 0);

    // 5) selective scan -> ys
    scan_kernel<<<DINNER / 256, 256, 0, stream>>>(delta, u, dbc, A_log, ys);

    // 6) ys = (ys + u*D) * silu(z)   (in-place)
    combine_kernel<<<SEQ * DINNER / 256, 256, 0, stream>>>(xz, u, D_skip, ys);

    // 7) o1 = ys @ out_proj^T                    (1024 x 2048, K=4096)
    wmma_gemm_tile<0, true><<<grd(DMODEL), blk, 0, stream>>>(
        ys, out_proj, o1, DMODEL, DINNER, DINNER, DINNER, DMODEL, nullptr, 0);

    // 8) out = o1 @ out_w^T + out_b, duplicated into both batch slices
    wmma_gemm_tile<2, true><<<grd(DOUTN), blk, 0, stream>>>(
        o1, out_w, out, DOUTN, DMODEL, DMODEL, DMODEL, DOUTN, out_b,
        (long long)SEQ * DOUTN);
}
